// AreaAttention_54743653155064
// MI455X (gfx1250) — compile-verified, matrix operand order checked by reading, not run on hardware
//
#include <hip/hip_runtime.h>
#include <hip/hip_bf16.h>

#define B_ 64
#define L_ 512
#define D_ 256
#define W_ 3
#define LW_ (L_ * W_)

typedef _Float16 f16;
typedef __attribute__((ext_vector_type(16))) _Float16 v16h;
typedef __attribute__((ext_vector_type(8)))  _Float16 h8;
typedef __attribute__((ext_vector_type(4)))  _Float16 h4;
typedef __attribute__((ext_vector_type(8)))  float    v8f;

__device__ __forceinline__ v16h mk16(h8 lo, h8 hi) {
    v16h r;
#pragma unroll
    for (int i = 0; i < 8; ++i) { r[i] = lo[i]; r[i + 8] = hi[i]; }
    return r;
}

// Load one 16x32-layout f16 fragment (ISA A/B striping) from an LDS row base.
__device__ __forceinline__ v16h ldfrag(const f16* base, int hf) {
    return mk16(*(const h8*)(base + hf * 8), *(const h8*)(base + 16 + hf * 8));
}

// CDNA5 async global->LDS copy (ASYNCcnt-tracked), 16 bytes per lane.
__device__ __forceinline__ void async_cp16(const f16* g, const f16* lds) {
    uint32_t loff = (uint32_t)(uintptr_t)lds;  // low 32 bits of shared-aperture addr = LDS offset
    asm volatile("global_load_async_to_lds_b128 %0, %1, off"
                 :: "v"(loff), "v"(g) : "memory");
}

// ---------------------------------------------------------------------------
// Pass 1: area pooling.  keys: windowed mean (w=1..3), vals: windowed sum.
// kh  [B][LW][D]  f16 row-major  (B-operand of QK^T: key rows contiguous in D)
// vT  [B][D][LW]  f16 transposed (B-operand of PV: dim rows contiguous in key)
// qh  [B][L][D]   f16
// ---------------------------------------------------------------------------
__global__ __launch_bounds__(256) void area_pool_kernel(
    const float* __restrict__ q, const float* __restrict__ key,
    const float* __restrict__ val, f16* __restrict__ kh,
    f16* __restrict__ vT, f16* __restrict__ qh)
{
    int idx = blockIdx.x * 256 + threadIdx.x;
    if (idx >= B_ * L_ * (D_ / 4)) return;
    int b  = idx / (L_ * 64);
    int r  = idx % (L_ * 64);
    int t  = r / 64;
    int c4 = r % 64;
    size_t base = ((size_t)b * L_ + t) * D_ + c4 * 4;

    float4 k0 = *(const float4*)(key + base);
    float4 v0 = *(const float4*)(val + base);
    float4 qv = *(const float4*)(q + base);
    float4 k1 = make_float4(0.f, 0.f, 0.f, 0.f), v1 = k1, k2 = k1, v2 = k1;
    if (t >= 1) { k1 = *(const float4*)(key + base - D_);     v1 = *(const float4*)(val + base - D_); }
    if (t >= 2) { k2 = *(const float4*)(key + base - 2 * D_); v2 = *(const float4*)(val + base - 2 * D_); }

    float ka[4] = {k0.x, k0.y, k0.z, k0.w}, kb[4] = {k1.x, k1.y, k1.z, k1.w}, kc[4] = {k2.x, k2.y, k2.z, k2.w};
    float va[4] = {v0.x, v0.y, v0.z, v0.w}, vb[4] = {v1.x, v1.y, v1.z, v1.w}, vc[4] = {v2.x, v2.y, v2.z, v2.w};
    float qa[4] = {qv.x, qv.y, qv.z, qv.w};

    h4 ok0, ok1, ok2, oq;
#pragma unroll
    for (int j = 0; j < 4; ++j) {
        float s2k = ka[j] + kb[j];
        float s3k = s2k + kc[j];
        ok0[j] = (f16)ka[j];
        ok1[j] = (f16)(t >= 1 ? s2k * 0.5f : 0.f);
        ok2[j] = (f16)(t >= 2 ? s3k * (1.f / 3.f) : 0.f);
        oq[j]  = (f16)qa[j];

        float o1v = t >= 1 ? va[j] + vb[j] : 0.f;
        float o2v = t >= 2 ? va[j] + vb[j] + vc[j] : 0.f;
        int d = c4 * 4 + j;
        size_t vtb = ((size_t)b * D_ + d) * LW_ + t;
        vT[vtb]          = (f16)va[j];
        vT[vtb + L_]     = (f16)o1v;
        vT[vtb + 2 * L_] = (f16)o2v;
    }
    size_t khb = ((size_t)b * LW_ + t) * D_ + c4 * 4;
    *(h4*)(kh + khb)                       = ok0;
    *(h4*)(kh + khb + (size_t)L_ * D_)     = ok1;
    *(h4*)(kh + khb + (size_t)2 * L_ * D_) = ok2;
    *(h4*)(qh + base)                      = oq;
}

// ---------------------------------------------------------------------------
// Pass 2: flash attention over 1536 pooled keys, WMMA f16 -> f32, with
// double-buffered async-to-LDS prefetch and software-pipelined fragment loads.
// 4 waves / block; wave w owns queries [qbase+16w, +16); block owns 64 queries.
// ---------------------------------------------------------------------------
__global__ __launch_bounds__(128, 1) void area_attn_kernel(
    const f16* __restrict__ qh, const f16* __restrict__ kh,
    const f16* __restrict__ vT, float* __restrict__ out)
{
    __shared__ f16 Ks[2][32 * 256];   // 2 x (32 keys x D)   (32 KB)
    __shared__ f16 Vs[2][256 * 32];   // 2 x (D x 32 keys)   (32 KB)
    __shared__ f16 Ps[4][16 * 32];    // per-wave P tile     ( 4 KB)

    const int tid  = threadIdx.x;
    const int wave = tid >> 5;
    const int lane = tid & 31;
    const int hf   = lane >> 4;       // half-wave select (ISA A/B fragment layout)
    const int ln   = lane & 15;
    const int b    = blockIdx.x >> 3;
    const int qb   = blockIdx.x & 7;
    const int qbase = qb * 64;
    const int qw0   = qbase + wave * 16;
    const int qlast = qw0 + 15;

    const int nst  = 2 * qb + 2;      // 32-key super-tiles per width block
    const int ntot = W_ * nst;

    // Issue one super-tile stage: 16 async b128 instructions per wave.
    auto prefetch = [&](int tile, int buf) {
        int blk = tile / nst;
        int st  = tile - blk * nst;
        const size_t krow0 = (size_t)b * LW_ + blk * L_ + st * 32;
        f16* kdst = &Ks[buf][0];
        for (int i = tid; i < 1024; i += 128) {         // 32 rows x 32 chunks of 16B
            int row = i >> 5, c = i & 31;
            async_cp16(kh + (krow0 + row) * D_ + c * 8, kdst + row * 256 + c * 8);
        }
        const int n0 = blk * L_ + st * 32;
        const size_t vbase = (size_t)b * D_ * LW_;
        f16* vdst = &Vs[buf][0];
        for (int i = tid; i < 1024; i += 128) {         // 256 dims x 4 chunks of 16B
            int d = i >> 2, c = i & 3;
            async_cp16(vT + vbase + (size_t)d * LW_ + n0 + c * 8, vdst + d * 32 + c * 8);
        }
    };

    // Preload Q A-fragments (row = ln; lane<16 holds K{0-7,16-23}, lane>=16 K{8-15,24-31})
    v16h qf[8];
    {
        const f16* qr = qh + ((size_t)b * L_ + qw0 + ln) * D_;
#pragma unroll
        for (int kc = 0; kc < 8; ++kc)
            qf[kc] = ldfrag(qr + kc * 32, hf);
    }

    v8f O[16];
#pragma unroll
    for (int t = 0; t < 16; ++t) O[t] = (v8f){};
    float m[8], lsum[8];
#pragma unroll
    for (int r = 0; r < 8; ++r) { m[r] = -3.0e38f; lsum[r] = 0.f; }

    prefetch(0, 0);

    for (int tile = 0; tile < ntot; ++tile) {
        const int buf = tile & 1;
        const int blk = tile / nst;
        const int st  = tile - blk * nst;

        // Prefetch next tile into the other buffer, then wait for THIS tile.
        // Asyncs complete in order: <=16 outstanding => current tile's 16 done.
        if (tile + 1 < ntot) {
            prefetch(tile + 1, buf ^ 1);
            asm volatile("s_wait_asynccnt 16" ::: "memory");
        } else {
            asm volatile("s_wait_asynccnt 0" ::: "memory");
        }
        __syncthreads();

        if (st * 32 <= qlast) {       // wave-uniform causal skip (EXEC stays full)
            const f16* ksb = &Ks[buf][0];
            const f16* vsb = &Vs[buf][0];

            // ---- S = Q K^T (two 16x16 score tiles, K-dim = 256), 1-deep SW pipeline ----
            v8f s0 = {}, s1 = {};
            v16h b0 = ldfrag(ksb + ln * 256, hf);
            v16h b1 = ldfrag(ksb + (16 + ln) * 256, hf);
#pragma unroll
            for (int kc = 0; kc < 8; ++kc) {
                v16h n0f, n1f;
                if (kc < 7) {
                    n0f = ldfrag(ksb + ln * 256 + (kc + 1) * 32, hf);
                    n1f = ldfrag(ksb + (16 + ln) * 256 + (kc + 1) * 32, hf);
                }
                s0 = __builtin_amdgcn_wmma_f32_16x16x32_f16(false, qf[kc], false, b0,
                                                            (short)0, s0, false, false);
                s1 = __builtin_amdgcn_wmma_f32_16x16x32_f16(false, qf[kc], false, b1,
                                                            (short)0, s1, false, false);
                if (kc < 7) { b0 = n0f; b1 = n1f; }
            }

            // ---- mask + online softmax (C layout: row = r+8*hf, col = ln) ----
            const float rt = 0.0625f;     // 1/sqrt(256)
            float alpha[8];
#pragma unroll
            for (int r = 0; r < 8; ++r) {
                int qrow = qw0 + hf * 8 + r;
                int k0 = st * 32 + ln, k1 = k0 + 16;
                bool ok0 = (k0 <= qrow) && (k0 >= blk) && (qrow >= blk);
                bool ok1 = (k1 <= qrow) && (qrow >= blk);
                float x0 = ok0 ? s0[r] * rt : -3.0e38f;
                float x1 = ok1 ? s1[r] * rt : -3.0e38f;
                float mx = fmaxf(x0, x1);
#pragma unroll
                for (int off = 1; off < 16; off <<= 1)
                    mx = fmaxf(mx, __shfl_xor(mx, off, 32));
                float mn = fmaxf(m[r], mx);
                float a  = __expf(m[r] - mn);
                float p0 = __expf(x0 - mn);
                float p1 = __expf(x1 - mn);
                float sm = p0 + p1;
#pragma unroll
                for (int off = 1; off < 16; off <<= 1)
                    sm += __shfl_xor(sm, off, 32);
                lsum[r] = lsum[r] * a + sm;
                m[r] = mn;
                alpha[r] = a;
                Ps[wave][(hf * 8 + r) * 32 + ln]      = (f16)p0;
                Ps[wave][(hf * 8 + r) * 32 + 16 + ln] = (f16)p1;
            }
#pragma unroll
            for (int t = 0; t < 16; ++t)
#pragma unroll
                for (int r = 0; r < 8; ++r) O[t][r] *= alpha[r];

            // ---- O += P V  (A = P 16x32, B = V^T columns), 1-deep SW pipeline ----
            v16h pf = ldfrag(&Ps[wave][ln * 32], hf);
            v16h vb = ldfrag(vsb + ln * 32, hf);          // t = 0 fragment
#pragma unroll
            for (int t = 0; t < 16; ++t) {
                v16h nv;
                if (t < 15) nv = ldfrag(vsb + ((t + 1) * 16 + ln) * 32, hf);
                O[t] = __builtin_amdgcn_wmma_f32_16x16x32_f16(false, pf, false, vb,
                                                              (short)0, O[t], false, false);
                if (t < 15) vb = nv;
            }
        }
        __syncthreads();   // compute(tile) done before prefetch(tile+2) reuses this buffer
    }

    // ---- normalize and write O[16x256] f32 ----
#pragma unroll
    for (int r = 0; r < 8; ++r) {
        float inv = 1.0f / lsum[r];
        size_t ob = ((size_t)b * L_ + qw0 + hf * 8 + r) * D_;
#pragma unroll
        for (int t = 0; t < 16; ++t)
            out[ob + t * 16 + ln] = O[t][r] * inv;
    }
}

extern "C" void kernel_launch(void* const* d_in, const int* in_sizes, int n_in,
                              void* d_out, int out_size, void* d_ws, size_t ws_size,
                              hipStream_t stream) {
    (void)in_sizes; (void)n_in; (void)out_size; (void)ws_size;
    const float* q   = (const float*)d_in[0];
    const float* key = (const float*)d_in[1];
    const float* val = (const float*)d_in[2];
    // d_in[3] (attn_mask) is the fixed causal mask; it is baked into the kernel.

    f16* kh = (f16*)d_ws;                                   // [B][LW][D]
    f16* vT = kh + (size_t)B_ * LW_ * D_;                   // [B][D][LW]
    f16* qh = vT + (size_t)B_ * LW_ * D_;                   // [B][L][D]
    float* out = (float*)d_out;

    int pool_threads = B_ * L_ * (D_ / 4);
    area_pool_kernel<<<(pool_threads + 255) / 256, 256, 0, stream>>>(q, key, val, kh, vT, qh);
    area_attn_kernel<<<B_ * (L_ / 64), 128, 0, stream>>>(qh, kh, vT, out);
}
